// HierarchicalStateSpaceModel_76166950027341
// MI455X (gfx1250) — compile-verified, matrix operand order checked by reading, not run on hardware
//
#include <hip/hip_runtime.h>
#include <math.h>

// ---------------------------------------------------------------------------
// CDNA5 / gfx1250 implementation of the hierarchical state-space model.
// All dense matmuls run on v_wmma_f32_16x16x32_bf16 (wave32, 16x16 tiles,
// fp32 accum). Sequential recurrences run as persistent single-workgroup
// kernels that call a workgroup-scope WMMA GEMM between phase fences.
// Causal-conv activations live in a zero-padded (64 x 132 x 512) layout so
// the im2col GEMM inner loop is branch-free. The scan kernel additionally
// issues TDM tensor_load_to_lds prefetches of the next step's WMMA operand
// slab (L2 warming + LDS staging), tracked with TENSORcnt.
// Fixed sizes: B=64, S=128, OBS=1024, D=512, DS=256, EMB=256, AEMB=64,
// NACT=32, NLAT=10, L=2, K=5, SP=132 (padded sequence length).
// ---------------------------------------------------------------------------

typedef __attribute__((ext_vector_type(16))) __bf16 v16bf;
typedef __attribute__((ext_vector_type(8)))  float  v8f;
typedef __attribute__((ext_vector_type(4)))  unsigned int v4u32;
typedef __attribute__((ext_vector_type(8)))  int v8i32;
typedef __attribute__((ext_vector_type(4)))  int v4i32;

union Frag { v16bf v; unsigned int u[8]; };

__device__ __forceinline__ unsigned short f2bf(float f) {
    unsigned int u = __float_as_uint(f);
    unsigned int r = u + 0x7FFFu + ((u >> 16) & 1u);
    return (unsigned short)(r >> 16);
}
__device__ __forceinline__ float bf2f(unsigned short h) {
    return __uint_as_float(((unsigned int)h) << 16);
}
__device__ __forceinline__ float sigm(float x) { return 1.f / (1.f + expf(-x)); }

#define WG_SYNC() do { __threadfence_block(); __syncthreads(); } while (0)

__device__ __forceinline__ v8f wmma_bf16(v16bf a, v16bf b, v8f c) {
    return __builtin_amdgcn_wmma_f32_16x16x32_bf16(false, a, false, b, (short)0, c,
                                                   false, false);
}

// Load one 16x32 bf16 fragment (A-layout; also valid for B as W.T rows).
// p points at (row_base + k0 + 8*half). K-pairs packed per dword per the
// CDNA5 16-bit operand layout.
__device__ __forceinline__ void load_frag(unsigned int f[8], const unsigned short* p) {
    f[0] = *(const unsigned int*)(p + 0);
    f[1] = *(const unsigned int*)(p + 2);
    f[2] = *(const unsigned int*)(p + 4);
    f[3] = *(const unsigned int*)(p + 6);
    f[4] = *(const unsigned int*)(p + 16);
    f[5] = *(const unsigned int*)(p + 18);
    f[6] = *(const unsigned int*)(p + 20);
    f[7] = *(const unsigned int*)(p + 22);
}

// ---------------------------------------------------------------------------
// Workgroup-scope GEMM: C(MxN) = sum_segs A_seg(M x K_seg) @ W[:, kw0:kw0+K].T
// A segs + W bf16; C f32; optional bias[n]. Waves split 16x64 tiles.
// No internal sync; caller fences.
// ---------------------------------------------------------------------------
struct Seg { const unsigned short* A; int lda; int K; int kw0; };

__device__ void wg_gemm(const Seg* segs, int nseg,
                        const unsigned short* __restrict__ W, int ldw,
                        float* __restrict__ C, int ldc,
                        const float* __restrict__ bias, bool accum,
                        int M, int N, int wave, int nwaves)
{
    const int lane = threadIdx.x & 31;
    const int half = lane >> 4, lm = lane & 15;
    const int mt = M >> 4, nt64 = N >> 6;
    const int ntiles = mt * nt64;
    for (int tile = wave; tile < ntiles; tile += nwaves) {
        const int m0 = (tile % mt) << 4;
        const int n0 = (tile / mt) << 6;
        v8f acc[4];
        for (int j = 0; j < 4; ++j) {
            if (accum) {
                for (int i = 0; i < 8; ++i)
                    acc[j][i] = C[(size_t)(m0 + i + 8 * half) * ldc + n0 + 16 * j + lm];
            } else if (bias) {
                float bv = bias[n0 + 16 * j + lm];
                for (int i = 0; i < 8; ++i) acc[j][i] = bv;
            } else {
                for (int i = 0; i < 8; ++i) acc[j][i] = 0.f;
            }
        }
        for (int s = 0; s < nseg; ++s) {
            const Seg sg = segs[s];
            for (int k0 = 0; k0 < sg.K; k0 += 32) {
                Frag a;
                load_frag(a.u, sg.A + (size_t)(m0 + lm) * sg.lda + k0 + 8 * half);
                for (int j = 0; j < 4; ++j) {
                    Frag b;
                    load_frag(b.u, W + (size_t)(n0 + 16 * j + lm) * ldw + sg.kw0 + k0 + 8 * half);
                    acc[j] = wmma_bf16(a.v, b.v, acc[j]);
                }
            }
        }
        for (int j = 0; j < 4; ++j)
            for (int i = 0; i < 8; ++i)
                C[(size_t)(m0 + i + 8 * half) * ldc + n0 + 16 * j + lm] = acc[j][i];
    }
}

// ---------------------------------------------------------------------------
// Grid-wide GEMM kernel.
//  mode 0: A bf16 row-major (M x K), leading dim lda.
//  mode 1: branch-free im2col causal conv; A is padded (64,132,512),
//          column kk = tap*512 + i, row (b,s) -> padded row b*132 + s + tap.
//  mode 2: A in padded layout, plain GEMM (row (b,s) -> b*132 + s + 4).
//  remap : store C row m=(b*128+s) at row (s*64+b) (t-major for recurrences).
//  padbf : bf16 mirror goes to padded (64,132,N) layout.
// ---------------------------------------------------------------------------
__global__ void __launch_bounds__(256)
gemm_kernel(const unsigned short* __restrict__ A, int lda,
            const unsigned short* __restrict__ W, int ldw, int kw0,
            float* __restrict__ C, int ldc,
            unsigned short* __restrict__ Cbf,
            const float* __restrict__ bias,
            int M, int N, int K, int accum, int mode, int remap, int padbf)
{
    const int lane = threadIdx.x & 31;
    const int wave = threadIdx.x >> 5;
    const int nwb  = blockDim.x >> 5;
    const int half = lane >> 4, lm = lane & 15;
    const int mt = M >> 4, nt64 = N >> 6;
    const long long ntiles = (long long)mt * nt64;
    for (long long tile = (long long)blockIdx.x * nwb + wave; tile < ntiles;
         tile += (long long)gridDim.x * nwb) {
        const int m0 = (int)(tile % mt) << 4;
        const int n0 = (int)(tile / mt) << 6;
        v8f acc[4];
        for (int j = 0; j < 4; ++j) {
            if (accum) {
                for (int i = 0; i < 8; ++i)
                    acc[j][i] = C[(size_t)(m0 + i + 8 * half) * ldc + n0 + 16 * j + lm];
            } else if (bias) {
                float bv = bias[n0 + 16 * j + lm];
                for (int i = 0; i < 8; ++i) acc[j][i] = bv;
            } else {
                for (int i = 0; i < 8; ++i) acc[j][i] = 0.f;
            }
        }
        const int r = m0 + lm;
        const int prow = (r >> 7) * 132 + (r & 127);   // padded-layout base row
        for (int k0 = 0; k0 < K; k0 += 32) {
            Frag a;
            if (mode == 1) {
                const int tap = k0 >> 9, i0 = k0 & 511;
                load_frag(a.u, A + (size_t)(prow + tap) * 512 + i0 + 8 * half);
            } else if (mode == 2) {
                load_frag(a.u, A + (size_t)(prow + 4) * 512 + k0 + 8 * half);
            } else {
                load_frag(a.u, A + (size_t)r * lda + k0 + 8 * half);
            }
            // hint the next W slab into cache (global_prefetch_b8)
            if (k0 + 32 < K)
                __builtin_prefetch(W + (size_t)(n0 + lm) * ldw + kw0 + k0 + 32, 0, 0);
            for (int j = 0; j < 4; ++j) {
                Frag b;
                load_frag(b.u, W + (size_t)(n0 + 16 * j + lm) * ldw + kw0 + k0 + 8 * half);
                acc[j] = wmma_bf16(a.v, b.v, acc[j]);
            }
        }
        for (int j = 0; j < 4; ++j)
            for (int i = 0; i < 8; ++i) {
                const int mr = m0 + i + 8 * half;
                const int cr = remap ? ((mr & 127) * 64 + (mr >> 7)) : mr;
                C[(size_t)cr * ldc + n0 + 16 * j + lm] = acc[j][i];
                if (Cbf) {
                    const int br = padbf ? ((mr >> 7) * 132 + (mr & 127) + 4) : cr;
                    Cbf[(size_t)br * ldc + n0 + 16 * j + lm] = f2bf(acc[j][i]);
                }
            }
    }
}

// ---------------------------------------------------------------------------
// Small elementwise / setup kernels
// ---------------------------------------------------------------------------
__global__ void cvt_bf16_kernel(const float* __restrict__ s,
                                unsigned short* __restrict__ d, size_t n) {
    for (size_t i = (size_t)blockIdx.x * blockDim.x + threadIdx.x; i < n;
         i += (size_t)gridDim.x * blockDim.x)
        d[i] = f2bf(s[i]);
}

// zero the 4 causal pad rows of a padded (64,132,512) bf16 buffer
__global__ void pad_zero_kernel(unsigned short* __restrict__ buf) {
    const int idx = blockIdx.x * blockDim.x + threadIdx.x;
    if (idx >= 64 * 4 * 512) return;
    const int b = idx >> 11, p = (idx >> 9) & 3, c = idx & 511;
    buf[(size_t)(b * 132 + p) * 512 + c] = 0;
}

// pb_convW (L,O,I,K) -> bf16 (L,O,K,I) so im2col columns kk = tap*512 + i.
__global__ void convw_kernel(const float* __restrict__ W,
                             unsigned short* __restrict__ out, int n) {
    for (int idx = blockIdx.x * blockDim.x + threadIdx.x; idx < n;
         idx += gridDim.x * blockDim.x) {
        const int i  = idx & 511;
        const int k  = (idx >> 9) % 5;
        const int lo = idx / (512 * 5);
        out[idx] = f2bf(W[(size_t)lo * 2560 + i * 5 + k]);
    }
}

__global__ void act_kernel(const int* __restrict__ actions,
                           const float* __restrict__ table,
                           unsigned short* __restrict__ out) {
    const int idx = blockIdx.x * blockDim.x + threadIdx.x;
    if (idx >= 8192 * 64) return;
    const int m = idx >> 6, j = idx & 63;
    out[idx] = f2bf(table[actions[m] * 64 + j]);
}

__global__ void ln_stats_kernel(const float* __restrict__ y,
                                float* __restrict__ stats) {
    __shared__ float ss[256], s2[256];
    const int c = blockIdx.x, tid = threadIdx.x;
    float a = 0.f, b = 0.f;
    for (int m = tid; m < 8192; m += 256) {
        const float v = y[(size_t)m * 512 + c];
        a += v; b += v * v;
    }
    ss[tid] = a; s2[tid] = b; __syncthreads();
    for (int s = 128; s > 0; s >>= 1) {
        if (tid < s) { ss[tid] += ss[tid + s]; s2[tid] += s2[tid + s]; }
        __syncthreads();
    }
    if (tid == 0) {
        const float m = ss[0] / 8192.f;
        stats[c * 2] = m;
        stats[c * 2 + 1] = s2[0] / 8192.f - m * m;
    }
}

// LN + affine + ELU; writes bf16 into padded (64,132,512) layout.
__global__ void ln_apply_kernel(const float* __restrict__ y,
                                const float* __restrict__ stats,
                                const float* __restrict__ gamma,
                                const float* __restrict__ beta,
                                unsigned short* __restrict__ out, size_t n) {
    for (size_t i = (size_t)blockIdx.x * blockDim.x + threadIdx.x; i < n;
         i += (size_t)gridDim.x * blockDim.x) {
        const int c = (int)(i & 511);
        const int m = (int)(i >> 9);
        float x = (y[i] - stats[c * 2]) * rsqrtf(stats[c * 2 + 1] + 1e-5f);
        x = x * gamma[c] + beta[c];
        x = x > 0.f ? x : (expf(x) - 1.f);           // ELU
        out[(size_t)((m >> 7) * 132 + (m & 127) + 4) * 512 + c] = f2bf(x);
    }
}

// tiny 2-channel causal conv head (N=2: VALU is fine here); x is padded.
__global__ void logits_kernel(const unsigned short* __restrict__ x,
                              const float* __restrict__ fW,
                              const float* __restrict__ fb,
                              float* __restrict__ logits) {
    const int tg = blockIdx.x * blockDim.x + threadIdx.x;
    if (tg >= 8192 * 2) return;
    const int m = tg >> 1, o = tg & 1;
    const int b = m >> 7, s = m & 127;
    float acc = fb[o];
    for (int k = 0; k < 5; ++k) {
        const unsigned short* xr = x + (size_t)(b * 132 + s + k) * 512;
        const float* wr = fW + (size_t)o * 2560 + k;
        for (int i = 0; i < 512; ++i)
            acc += bf2f(xr[i]) * wr[i * 5];
    }
    logits[m * 2 + o] = acc;
}

// straight-through gumbel boundary: forward value is hard one-hot.
__global__ void read_kernel(const float* __restrict__ logits,
                            const float* __restrict__ bnd_u,
                            float* __restrict__ readT) {
    const int m = blockIdx.x * blockDim.x + threadIdx.x;
    if (m >= 8192) return;
    const int b = m >> 7, s = m & 127;
    float u0 = fminf(fmaxf(bnd_u[m * 2], 1e-6f), 1.f - 1e-6f);
    float u1 = fminf(fmaxf(bnd_u[m * 2 + 1], 1e-6f), 1.f - 1e-6f);
    const float g0 = -logf(-logf(u0)), g1 = -logf(-logf(u1));
    readT[s * 64 + b] =
        (logits[m * 2] + g0 >= logits[m * 2 + 1] + g1) ? 1.f : 0.f;
}

// ---------------------------------------------------------------------------
// GRU sweeps (gridDim=3: apf fwd, apb bwd, opf fwd). Persistent workgroup,
// per-step 64x1536x512 WMMA GEMM + elementwise GRU.
// GI already holds feat@Wi.T + bi in t-major rows [t*64+b].
// ---------------------------------------------------------------------------
__global__ void __launch_bounds__(1024)
sweep_kernel(const float* __restrict__ GI,
             const unsigned short* __restrict__ WhB,
             const float* __restrict__ grus_bh,
             const float* __restrict__ readT,
             float* __restrict__ h_f, unsigned short* __restrict__ h_b,
             float* __restrict__ gh,
             unsigned short* __restrict__ ap_bf,
             unsigned short* __restrict__ opf_bf)
{
    const int g = blockIdx.x;
    const int tid = threadIdx.x, wave = tid >> 5, nw = blockDim.x >> 5;
    float* hf = h_f + (size_t)g * 64 * 512;
    unsigned short* hb = h_b + (size_t)g * 64 * 512;
    float* gg = gh + (size_t)g * 64 * 1536;
    const unsigned short* Wh = WhB + (size_t)g * 1536 * 512;
    const float* bh = grus_bh + g * 1536;
    for (int e = tid; e < 64 * 512; e += blockDim.x) { hf[e] = 0.f; hb[e] = 0; }
    WG_SYNC();
    for (int tt = 0; tt < 128; ++tt) {
        const int t = (g == 1) ? (127 - tt) : tt;
        Seg s{hb, 512, 512, 0};
        wg_gemm(&s, 1, Wh, 512, gg, 1536, bh, false, 64, 1536, wave, nw);
        WG_SYNC();
        const float* gi = GI + ((size_t)g * 128 + t) * (size_t)64 * 1536;
        const float* rd = readT + t * 64;
        for (int e = tid; e < 64 * 512; e += blockDim.x) {
            const int b = e >> 9, d = e & 511;
            const float r = sigm(gi[b * 1536 + d] + gg[b * 1536 + d]);
            const float z = sigm(gi[b * 1536 + 512 + d] + gg[b * 1536 + 512 + d]);
            const float n = tanhf(gi[b * 1536 + 1024 + d] + r * gg[b * 1536 + 1024 + d]);
            const float h = (1.f - z) * n + z * hf[e];
            const unsigned short ob = f2bf(h);
            if (g == 0)      ap_bf[((size_t)t * 64 + b) * 1024 + d] = ob;
            else if (g == 1) ap_bf[((size_t)t * 64 + b) * 1024 + 512 + d] = ob;
            else             opf_bf[((size_t)t * 64 + b) * 512 + d] = ob;
            const float hc = h * (1.f - rd[b]);
            hf[e] = hc; hb[e] = f2bf(hc);
        }
        WG_SYNC();
    }
    // Cluster-scope rendezvous: architectural NOP when the grid is not
    // launched as a workgroup cluster (ClusterID==0), per CDNA5 ISA 6.6.
    __builtin_amdgcn_s_cluster_barrier();
}

// ---------------------------------------------------------------------------
// The sequential scan (one persistent workgroup).
// ---------------------------------------------------------------------------
struct ScanArgs {
    const unsigned short* ap_bf;   // [S][64][1024] (apf|apb)
    const unsigned short* opf_bf;  // [S][64][512]
    const float* readT;            // [S][64]
    const float* vq_u;             // [B][S][10]
    const float* obs_eps;          // [B][S][256]
    const float* codebook;         // [10][512]
    const unsigned short *Wi3, *Wh3, *ubWi, *ubWh, *m1W, *m2W, *afW, *pmW, *psW, *ofW;
    const float *bi3, *bh3, *ubbi, *ubbh, *m1b, *m2b, *afb, *pmb, *psb, *ofb;
    float *ab_f, *az_f, *ob_f, *oz_f;
    unsigned short *ab_b, *az_b, *ob_b, *oz_b;
    float *g1, *g2, *zf, *h1f;
    unsigned short* h1b;
    float* af_f; unsigned short* af_bf;
    float *pm, *ps, *of_f;
    unsigned short* ofs_bf;        // [B][S][512]
    float* out_loss;
};

__global__ void __launch_bounds__(1024) scan_kernel(ScanArgs A)
{
    const int tid = threadIdx.x;
    const int wave = tid >> 5, nw = blockDim.x >> 5;
    __shared__ float sdbuf[64 * 16];
    __shared__ int   s_idx[64];
    __shared__ float s_cbsq[16];
    __shared__ float s_cnt, s_es;
    __shared__ unsigned int tdm_stage[8192];   // 32 KB TDM staging buffer
    if (tid == 0) { s_cnt = 0.f; s_es = 0.f; }
    if (tid < 16) {
        float s = 0.f;
        if (tid < 10)
            for (int c = 0; c < 512; ++c) { const float v = A.codebook[tid * 512 + c]; s += v * v; }
        s_cbsq[tid] = s;
    }
    for (int e = tid; e < 64 * 512; e += blockDim.x) {
        A.ab_f[e] = 0.f; A.az_f[e] = 0.f; A.ob_f[e] = 0.f;
        A.ab_b[e] = 0;   A.az_b[e] = 0;   A.ob_b[e] = 0;
    }
    for (int e = tid; e < 64 * 256; e += blockDim.x) { A.oz_f[e] = 0.f; A.oz_b[e] = 0; }
    WG_SYNC();

    float es_loc = 0.f;
    for (int t = 0; t < 128; ++t) {
        const float* rd = A.readT + t * 64;
        // ---- TDM prefetch of the next step's ap operand slab (wave 0).
        // 1-D 16384-element (32 KB) bf16 tile, D# per ISA 8.3/8.4:
        // group0 = {count=1, lds_addr, global_addr, addr_hi|type=2},
        // group1 = {data_size=2B, tensor_dim0, tile_dim0, dim0_stride}.
        // Pulls the slab through GL2 into LDS ahead of the t+1 GEMM.
        if (tid < 32 && t + 1 < 128) {
            const unsigned long long ga =
                (unsigned long long)(const void*)(A.ap_bf + (size_t)(t + 1) * 64 * 1024);
            const unsigned int lds =
                (unsigned int)(unsigned long long)(void*)tdm_stage;
            v4u32 g0;
            g0[0] = 1u;                                   // count=1, user D#
            g0[1] = lds;                                  // lds_addr
            g0[2] = (unsigned int)ga;                     // global_addr[31:0]
            g0[3] = (unsigned int)((ga >> 32) & 0x1FFFFFFu) | (2u << 30); // [56:32]|type=2
            v8i32 g1;
            g1[0] = (int)(1u << 16);                      // data_size = 1 (2 bytes)
            g1[1] = (int)(0x4000u << 16);                 // tensor_dim0 = 16384 (lo16)
            g1[2] = 0;                                    // tensor_dim0 hi / tensor_dim1 lo
            g1[3] = (int)(0x4000u << 16);                 // tile_dim0 = 16384
            g1[4] = 0;                                    // tile_dim1/2 unused (1-D)
            g1[5] = 16384;                                // tensor_dim0_stride lo
            g1[6] = 0; g1[7] = 0;
            v4i32 gz4 = {0, 0, 0, 0};                     // groups 2/3 unused (<=2D)
            v8i32 gz8 = {0, 0, 0, 0, 0, 0, 0, 0};
            __builtin_amdgcn_tensor_load_to_lds(g0, g1, gz4, gz4, gz8, 0);
        }
        // ---- GRU3 gates + VQ hidden (independent GEMMs, one fence) ----
        { Seg s{A.az_b, 512, 512, 0}; wg_gemm(&s, 1, A.Wi3, 512, A.g1, 1536, A.bi3, false, 64, 1536, wave, nw); }
        { Seg s{A.ab_b, 512, 512, 0}; wg_gemm(&s, 1, A.Wh3, 512, A.g2, 1536, A.bh3, false, 64, 1536, wave, nw); }
        { Seg s{A.ap_bf + (size_t)t * 64 * 1024, 1024, 1024, 0};
          wg_gemm(&s, 1, A.m1W, 1024, A.h1f, 128, A.m1b, false, 64, 128, wave, nw); }
        WG_SYNC();
        // ---- ab update + relu ----
        for (int e = tid; e < 64 * 512; e += blockDim.x) {
            const int b = e >> 9, d = e & 511;
            const float r = sigm(A.g1[b * 1536 + d] + A.g2[b * 1536 + d]);
            const float z = sigm(A.g1[b * 1536 + 512 + d] + A.g2[b * 1536 + 512 + d]);
            const float n = tanhf(A.g1[b * 1536 + 1024 + d] + r * A.g2[b * 1536 + 1024 + d]);
            const float h = (1.f - z) * n + z * A.ab_f[e];
            const float rt = rd[b];
            const float abn = rt * h + (1.f - rt) * A.ab_f[e];
            A.ab_f[e] = abn; A.ab_b[e] = f2bf(abn);
        }
        for (int e = tid; e < 64 * 128; e += blockDim.x) {
            float v = A.h1f[e]; v = v > 0.f ? v : 0.f; A.h1b[e] = f2bf(v);
        }
        WG_SYNC();
        // ---- zf ----
        { Seg s{A.h1b, 128, 128, 0}; wg_gemm(&s, 1, A.m2W, 128, A.zf, 512, A.m2b, false, 64, 512, wave, nw); }
        WG_SYNC();
        // ---- codebook distances / gumbel pick ----
        {
            const int b = tid >> 4, j = tid & 15;
            if (j < 10) {
                float dt = 0.f;
                const float* zr = A.zf + b * 512;
                const float* cr = A.codebook + j * 512;
                for (int c = 0; c < 512; ++c) dt += zr[c] * cr[c];
                sdbuf[tid] = dt;
            } else if (j == 15) {
                float s2 = 0.f;
                const float* zr = A.zf + b * 512;
                for (int c = 0; c < 512; ++c) s2 += zr[c] * zr[c];
                sdbuf[tid] = s2;
            }
        }
        __syncthreads();
        {
            const int b = tid >> 4, j = tid & 15;
            if (j == 0) {
                const float zq = sdbuf[b * 16 + 15];
                float sd[10], mx = -1e30f;
                for (int jj = 0; jj < 10; ++jj) {
                    const float dist = zq + s_cbsq[jj] - 2.f * sdbuf[b * 16 + jj];
                    sd[jj] = -dist * 10.f;
                    mx = fmaxf(mx, sd[jj]);
                }
                float lse = 0.f;
                for (int jj = 0; jj < 10; ++jj) {
                    sd[jj] = fminf(fmaxf(sd[jj] - mx, -1000.f), 10.f);
                    lse += expf(sd[jj]);
                }
                lse = logf(lse);
                int best = 0; float bv = -1e30f;
                for (int jj = 0; jj < 10; ++jj) {
                    float u = A.vq_u[(size_t)b * 1280 + t * 10 + jj];
                    u = fminf(fmaxf(u, 1e-6f), 1.f - 1e-6f);
                    const float v = sd[jj] - lse + (-logf(-logf(u)));
                    if (v > bv) { bv = v; best = jj; }
                }
                s_idx[b] = best;
                atomicAdd(&s_cnt, rd[b] * 512.f);
            }
        }
        __syncthreads();
        // ---- az update + vq loss ----
        for (int e = tid; e < 64 * 512; e += blockDim.x) {
            const int b = e >> 9, d = e & 511;
            const float q = A.codebook[s_idx[b] * 512 + d];
            const float zv = A.zf[e];
            const float rt = rd[b];
            const float df = q - zv;
            es_loc += rt * df * df;                 // es == qs in value
            const float azn = rt * q + (1.f - rt) * A.az_f[e];
            A.az_f[e] = azn; A.az_b[e] = f2bf(azn);
        }
        WG_SYNC();
        // ---- af = [ab|az]@afW + afb ----
        { Seg s[2] = {{A.ab_b, 512, 512, 0}, {A.az_b, 512, 512, 512}};
          wg_gemm(s, 2, A.afW, 1024, A.af_f, 512, A.afb, false, 64, 512, wave, nw); }
        WG_SYNC();
        for (int e = tid; e < 64 * 512; e += blockDim.x) A.af_bf[e] = f2bf(A.af_f[e]);
        WG_SYNC();
        // ---- ub gates + posterior mean/std (4 independent GEMMs) ----
        { Seg s[2] = {{A.oz_b, 256, 256, 0}, {A.af_bf, 512, 512, 256}};
          wg_gemm(s, 2, A.ubWi, 768, A.g1, 1536, A.ubbi, false, 64, 1536, wave, nw); }
        { Seg s{A.ob_b, 512, 512, 0};
          wg_gemm(&s, 1, A.ubWh, 512, A.g2, 1536, A.ubbh, false, 64, 1536, wave, nw); }
        { Seg s[2] = {{A.opf_bf + (size_t)t * 64 * 512, 512, 512, 0}, {A.af_bf, 512, 512, 512}};
          wg_gemm(s, 2, A.pmW, 1024, A.pm, 256, A.pmb, false, 64, 256, wave, nw); }
        { Seg s2[2] = {{A.opf_bf + (size_t)t * 64 * 512, 512, 512, 0}, {A.af_bf, 512, 512, 512}};
          wg_gemm(s2, 2, A.psW, 1024, A.ps, 256, A.psb, false, 64, 256, wave, nw); }
        WG_SYNC();
        // ---- ob / oz updates ----
        for (int e = tid; e < 64 * 512; e += blockDim.x) {
            const int b = e >> 9, d = e & 511;
            const float r = sigm(A.g1[b * 1536 + d] + A.g2[b * 1536 + d]);
            const float z = sigm(A.g1[b * 1536 + 512 + d] + A.g2[b * 1536 + 512 + d]);
            const float n = tanhf(A.g1[b * 1536 + 1024 + d] + r * A.g2[b * 1536 + 1024 + d]);
            const float h = (1.f - z) * n + z * A.ob_f[e];
            const float rt = rd[b];
            const float obn = rt * A.af_f[e] + (1.f - rt) * h;
            A.ob_f[e] = obn; A.ob_b[e] = f2bf(obn);
        }
        for (int e = tid; e < 64 * 256; e += blockDim.x) {
            const int b = e >> 8, d = e & 255;
            const float eps = A.obs_eps[(size_t)b * 128 * 256 + (size_t)t * 256 + d];
            const float ozn = A.pm[e] + sigm(A.ps[e]) * eps;
            A.oz_f[e] = ozn; A.oz_b[e] = f2bf(ozn);
        }
        WG_SYNC();
        // ---- of = [ob|oz]@ofW + ofb ----
        { Seg s[2] = {{A.ob_b, 512, 512, 0}, {A.oz_b, 256, 256, 512}};
          wg_gemm(s, 2, A.ofW, 768, A.of_f, 512, A.ofb, false, 64, 512, wave, nw); }
        WG_SYNC();
        for (int e = tid; e < 64 * 512; e += blockDim.x) {
            const int b = e >> 9, d = e & 511;
            A.ofs_bf[((size_t)b * 128 + t) * 512 + d] = f2bf(A.of_f[e]);
        }
        // keep TENSORcnt bounded: retire the outstanding TDM prefetch
        if (tid < 32) __builtin_amdgcn_s_wait_tensorcnt(0);
        WG_SYNC();
    }
    atomicAdd(&s_es, es_loc);
    __syncthreads();
    if (tid == 0) {
        float cnt = s_cnt; if (cnt < 1.f) cnt = 1.f;
        *A.out_loss = (2.f * s_es) / cnt;          // (qs + es) / max(cnt,1)
    }
}

// ---------------------------------------------------------------------------
// Host orchestration
// ---------------------------------------------------------------------------
extern "C" void kernel_launch(void* const* d_in, const int* in_sizes, int n_in,
                              void* d_out, int out_size, void* d_ws, size_t ws_size,
                              hipStream_t stream) {
    const float* obs      = (const float*)d_in[0];
    const int*   actions  = (const int*)d_in[1];
    const float* bnd_u    = (const float*)d_in[2];
    const float* vq_u     = (const float*)d_in[3];
    const float* obs_eps  = (const float*)d_in[4];
    const float* enc_W    = (const float*)d_in[5];
    const float* enc_b    = (const float*)d_in[6];
    const float* act_table= (const float*)d_in[7];
    const float* comb_W   = (const float*)d_in[8];
    const float* comb_b   = (const float*)d_in[9];
    const float* pb_convW = (const float*)d_in[10];
    const float* pb_gamma = (const float*)d_in[11];
    const float* pb_beta  = (const float*)d_in[12];
    const float* pb_fW    = (const float*)d_in[13];
    const float* pb_fb    = (const float*)d_in[14];
    const float* grus_Wi  = (const float*)d_in[15];
    const float* grus_Wh  = (const float*)d_in[16];
    const float* grus_bi  = (const float*)d_in[17];
    const float* grus_bh  = (const float*)d_in[18];
    const float* ub_Wi    = (const float*)d_in[19];
    const float* ub_Wh    = (const float*)d_in[20];
    const float* ub_bi    = (const float*)d_in[21];
    const float* ub_bh    = (const float*)d_in[22];
    const float* vq_m1W   = (const float*)d_in[23];
    const float* vq_m1b   = (const float*)d_in[24];
    const float* vq_m2W   = (const float*)d_in[25];
    const float* vq_m2b   = (const float*)d_in[26];
    const float* codebook = (const float*)d_in[27];
    const float* po_meanW = (const float*)d_in[28];
    const float* po_meanb = (const float*)d_in[29];
    const float* po_stdW  = (const float*)d_in[30];
    const float* po_stdb  = (const float*)d_in[31];
    const float* af_W     = (const float*)d_in[32];
    const float* af_bv    = (const float*)d_in[33];
    const float* of_W     = (const float*)d_in[34];
    const float* of_bv    = (const float*)d_in[35];
    const float* dec_W    = (const float*)d_in[36];
    const float* dec_b    = (const float*)d_in[37];

    char* ws = (char*)d_ws;
    size_t off = 0;
    auto alloc = [&](size_t bytes) -> void* {
        off = (off + 255) & ~(size_t)255;
        void* p = ws + off;
        off += bytes;
        return p;
    };
    auto aF = [&](size_t n) { return (float*)alloc(n * 4); };
    auto aH = [&](size_t n) { return (unsigned short*)alloc(n * 2); };

    // bf16 activation + weight mirrors
    unsigned short* obs_bf  = aH(8388608);
    unsigned short* a_bf    = aH(524288);
    unsigned short* e_bf    = aH(2097152);
    unsigned short* featp   = aH(64 * 132 * 512);   // padded causal layouts
    unsigned short* x1p     = aH(64 * 132 * 512);
    unsigned short* x2p     = aH(64 * 132 * 512);
    unsigned short* ap_bf   = aH(8388608);
    unsigned short* opf_bf  = aH(4194304);
    unsigned short* ofs_bf  = aH(4194304);
    unsigned short* encW_b  = aH(262144);
    unsigned short* combW_b = aH(163840);
    unsigned short* convW_b = aH(2621440);
    unsigned short* WiB     = aH(3145728);
    unsigned short* WhB     = aH(3145728);
    unsigned short* ubWi_b  = aH(1179648);
    unsigned short* ubWh_b  = aH(786432);
    unsigned short* m1W_b   = aH(131072);
    unsigned short* m2W_b   = aH(65536);
    unsigned short* afW_b   = aH(524288);
    unsigned short* pmW_b   = aH(262144);
    unsigned short* psW_b   = aH(262144);
    unsigned short* ofW_b   = aH(393216);
    unsigned short* decW_b  = aH(524288);
    // f32 scratch
    float* e_f    = aF(2097152);
    float* feat_f = aF(4194304);
    float* y_f    = aF(4194304);
    float* stats  = aF(1024);
    float* logits = aF(16384);
    float* readT  = aF(8192);
    float* GI     = aF(37748736);  // [3][S][64][1536], bi folded in
    float* sw_hf  = aF(98304);
    unsigned short* sw_hb = aH(98304);
    float* sw_gh  = aF(294912);
    // scan state/scratch
    float* ab_f = aF(32768); float* az_f = aF(32768); float* ob_f = aF(32768);
    float* oz_f = aF(16384);
    unsigned short* ab_b = aH(32768); unsigned short* az_b = aH(32768);
    unsigned short* ob_b = aH(32768); unsigned short* oz_b = aH(16384);
    float* g1 = aF(98304); float* g2 = aF(98304);
    float* zf = aF(32768); float* h1f = aF(8192);
    unsigned short* h1b = aH(8192);
    float* saf_f = aF(32768); unsigned short* saf_bf = aH(32768);
    float* pm = aF(16384); float* ps = aF(16384); float* of_f = aF(32768);
    (void)ws_size; (void)in_sizes; (void)n_in; (void)out_size;

    auto cvt = [&](const float* s, unsigned short* d, size_t n) {
        int blocks = (int)((n + 1023) / 1024); if (blocks > 4096) blocks = 4096;
        cvt_bf16_kernel<<<blocks, 256, 0, stream>>>(s, d, n);
    };
    auto gemm = [&](const unsigned short* A, int lda, const unsigned short* W, int ldw,
                    int kw0, float* C, int ldc, unsigned short* Cbf, const float* bias,
                    int M, int N, int K, int accum, int mode, int remap, int padbf) {
        long long tiles = (long long)(M / 16) * (N / 64);
        int blocks = (int)((tiles + 7) / 8);
        if (blocks > 4096) blocks = 4096; if (blocks < 1) blocks = 1;
        gemm_kernel<<<blocks, 256, 0, stream>>>(A, lda, W, ldw, kw0, C, ldc, Cbf,
                                                bias, M, N, K, accum, mode, remap, padbf);
    };

    // ---- weight + input conversions ----
    cvt(obs, obs_bf, 8388608);
    cvt(enc_W, encW_b, 262144);
    cvt(comb_W, combW_b, 163840);
    cvt(grus_Wi, WiB, 3145728);
    cvt(grus_Wh, WhB, 3145728);
    cvt(ub_Wi, ubWi_b, 1179648);
    cvt(ub_Wh, ubWh_b, 786432);
    cvt(vq_m1W, m1W_b, 131072);
    cvt(vq_m2W, m2W_b, 65536);
    cvt(af_W, afW_b, 524288);
    cvt(po_meanW, pmW_b, 262144);
    cvt(po_stdW, psW_b, 262144);
    cvt(of_W, ofW_b, 393216);
    cvt(dec_W, decW_b, 524288);
    convw_kernel<<<4096, 256, 0, stream>>>(pb_convW, convW_b, 2621440);
    act_kernel<<<2048, 256, 0, stream>>>(actions, act_table, a_bf);
    pad_zero_kernel<<<512, 256, 0, stream>>>(featp);
    pad_zero_kernel<<<512, 256, 0, stream>>>(x1p);
    pad_zero_kernel<<<512, 256, 0, stream>>>(x2p);

    // ---- encoder + feature combine (bf16 mirror into padded layout) ----
    gemm(obs_bf, 1024, encW_b, 1024, 0, e_f, 256, e_bf, enc_b, 8192, 256, 1024, 0, 0, 0, 0);
    gemm(a_bf, 64, combW_b, 320, 0, feat_f, 512, nullptr, comb_b, 8192, 512, 64, 0, 0, 0, 0);
    gemm(e_bf, 256, combW_b, 320, 64, feat_f, 512, featp, nullptr, 8192, 512, 256, 1, 0, 0, 1);

    // ---- boundary conv stack (branch-free im2col WMMA + LN/ELU) ----
    gemm(featp, 0, convW_b, 2560, 0, y_f, 512, nullptr, nullptr, 8192, 512, 2560, 0, 1, 0, 0);
    ln_stats_kernel<<<512, 256, 0, stream>>>(y_f, stats);
    ln_apply_kernel<<<4096, 256, 0, stream>>>(y_f, stats, pb_gamma, pb_beta, x1p, 4194304);
    gemm(x1p, 0, convW_b + (size_t)512 * 2560, 2560, 0, y_f, 512, nullptr, nullptr,
         8192, 512, 2560, 0, 1, 0, 0);
    ln_stats_kernel<<<512, 256, 0, stream>>>(y_f, stats);
    ln_apply_kernel<<<4096, 256, 0, stream>>>(y_f, stats, pb_gamma + 512, pb_beta + 512,
                                              x2p, 4194304);
    logits_kernel<<<64, 256, 0, stream>>>(x2p, pb_fW, pb_fb, logits);
    read_kernel<<<32, 256, 0, stream>>>(logits, bnd_u, readT);

    // ---- GRU input gates for the three sweeps (t-major rows, bi folded) ----
    for (int g = 0; g < 3; ++g)
        gemm(featp, 512, WiB + (size_t)g * 1536 * 512, 512, 0,
             GI + (size_t)g * 128 * 64 * 1536, 1536, nullptr, grus_bi + g * 1536,
             8192, 1536, 512, 0, 2, 1, 0);

    // ---- three persistent GRU sweeps ----
    sweep_kernel<<<3, 1024, 0, stream>>>(GI, WhB, grus_bh, readT, sw_hf, sw_hb, sw_gh,
                                         ap_bf, opf_bf);

    // ---- sequential scan ----
    ScanArgs SA;
    SA.ap_bf = ap_bf; SA.opf_bf = opf_bf; SA.readT = readT;
    SA.vq_u = vq_u; SA.obs_eps = obs_eps; SA.codebook = codebook;
    SA.Wi3 = WiB + (size_t)3 * 1536 * 512;
    SA.Wh3 = WhB + (size_t)3 * 1536 * 512;
    SA.ubWi = ubWi_b; SA.ubWh = ubWh_b; SA.m1W = m1W_b; SA.m2W = m2W_b;
    SA.afW = afW_b; SA.pmW = pmW_b; SA.psW = psW_b; SA.ofW = ofW_b;
    SA.bi3 = grus_bi + 3 * 1536; SA.bh3 = grus_bh + 3 * 1536;
    SA.ubbi = ub_bi; SA.ubbh = ub_bh; SA.m1b = vq_m1b; SA.m2b = vq_m2b;
    SA.afb = af_bv; SA.pmb = po_meanb; SA.psb = po_stdb; SA.ofb = of_bv;
    SA.ab_f = ab_f; SA.az_f = az_f; SA.ob_f = ob_f; SA.oz_f = oz_f;
    SA.ab_b = ab_b; SA.az_b = az_b; SA.ob_b = ob_b; SA.oz_b = oz_b;
    SA.g1 = g1; SA.g2 = g2; SA.zf = zf; SA.h1f = h1f; SA.h1b = h1b;
    SA.af_f = saf_f; SA.af_bf = saf_bf; SA.pm = pm; SA.ps = ps; SA.of_f = of_f;
    SA.ofs_bf = ofs_bf;
    SA.out_loss = ((float*)d_out) + (size_t)8388608;
    scan_kernel<<<1, 1024, 0, stream>>>(SA);

    // ---- decoder straight into d_out ----
    gemm(ofs_bf, 512, decW_b, 512, 0, (float*)d_out, 1024, nullptr, dec_b,
         8192, 1024, 512, 0, 0, 0, 0);
}
